// box_feature_extractor_69526930588029
// MI455X (gfx1250) — matrix operand order repaired and checked by reading, not compiled, not run on hardware
//
#include <hip/hip_runtime.h>
#include <stdint.h>

typedef __attribute__((ext_vector_type(16))) _Float16 v16h;
typedef __attribute__((ext_vector_type(8)))  _Float16 v8h;
typedef __attribute__((ext_vector_type(8)))  float    v8f;

#define BN_EPS 1e-5f

// ---------------------------------------------------------------------------
// Farthest point sampling: one block per batch, dist kept in global scratch.
// Emits carry-before-update index like the jax scan (idx[0] == 0).
// ---------------------------------------------------------------------------
__global__ void fps_kernel(const float* __restrict__ xyz, int* __restrict__ outIdx,
                           float* __restrict__ dist, int N, int npoint) {
  int b = blockIdx.x;
  const float* p = xyz + (size_t)b * N * 3;
  float* dst = dist + (size_t)b * N;
  __shared__ float redV[256];
  __shared__ int   redI[256];
  __shared__ int   sFar;
  for (int n = threadIdx.x; n < N; n += blockDim.x) dst[n] = 1e10f;
  if (threadIdx.x == 0) sFar = 0;
  __syncthreads();
  for (int i = 0; i < npoint; ++i) {
    int far = sFar;
    if (threadIdx.x == 0) outIdx[(size_t)b * npoint + i] = far;
    float cx = p[far*3+0], cy = p[far*3+1], cz = p[far*3+2];
    float best = -1.0f; int bestI = 0x7fffffff;
    for (int n = threadIdx.x; n < N; n += blockDim.x) {
      float dx = p[n*3+0]-cx, dy = p[n*3+1]-cy, dz = p[n*3+2]-cz;
      float d = dx*dx + dy*dy + dz*dz;
      float dn = fminf(dst[n], d);
      dst[n] = dn;
      if (dn > best || (dn == best && n < bestI)) { best = dn; bestI = n; }
    }
    redV[threadIdx.x] = best; redI[threadIdx.x] = bestI;
    __syncthreads();
    for (int s = 128; s > 0; s >>= 1) {
      if (threadIdx.x < s) {
        float v2 = redV[threadIdx.x + s]; int i2 = redI[threadIdx.x + s];
        if (v2 > redV[threadIdx.x] ||
            (v2 == redV[threadIdx.x] && i2 < redI[threadIdx.x])) {
          redV[threadIdx.x] = v2; redI[threadIdx.x] = i2;
        }
      }
      __syncthreads();
    }
    if (threadIdx.x == 0) sFar = redI[0];
    __syncthreads();
  }
}

// ---------------------------------------------------------------------------
__global__ void gather_xyz_kernel(const float* __restrict__ xyz, const int* __restrict__ idx,
                                  float* __restrict__ out, int total, int N, int S) {
  int i = blockIdx.x * blockDim.x + threadIdx.x;
  if (i >= total) return;
  int b = i / S;
  int n = idx[i];
  const float* src = xyz + ((size_t)b * N + n) * 3;
  out[(size_t)i*3+0] = src[0];
  out[(size_t)i*3+1] = src[1];
  out[(size_t)i*3+2] = src[2];
}

// ---------------------------------------------------------------------------
// Ball query: one wave32 per center; ballot pops hits in index order, pads
// short lists with the first hit (matches reference semantics).
// ---------------------------------------------------------------------------
__global__ void ball_query_kernel(const float* __restrict__ xyz, const float* __restrict__ cen,
                                  int* __restrict__ idx, int BS, int N, int S, int K, float r2) {
  int center = blockIdx.x * blockDim.y + threadIdx.y;
  if (center >= BS) return;
  int b = center / S;
  const float* p = xyz + (size_t)b * N * 3;
  float cx = cen[(size_t)center*3+0];
  float cy = cen[(size_t)center*3+1];
  float cz = cen[(size_t)center*3+2];
  int lane = threadIdx.x;
  int cnt = 0, first = -1;
  int* out = idx + (size_t)center * K;
  for (int base = 0; base < N && cnt < K; base += 32) {
    int n = base + lane;
    bool hit = false;
    if (n < N) {
      float dx = p[n*3+0]-cx, dy = p[n*3+1]-cy, dz = p[n*3+2]-cz;
      hit = (dx*dx + dy*dy + dz*dz) < r2;
    }
#if __has_builtin(__builtin_amdgcn_ballot_w32)
    unsigned mask = __builtin_amdgcn_ballot_w32(hit);
#else
    unsigned mask = (unsigned)__ballot(hit);
#endif
    while (mask && cnt < K) {
      int j = __ffs(mask) - 1;
      mask &= mask - 1;
      int nn = base + j;
      if (first < 0) first = nn;
      if (lane == 0) out[cnt] = nn;
      ++cnt;
    }
  }
  if (first < 0) first = 0;
  if (lane == 0) for (int k = cnt; k < K; ++k) out[k] = first;
}

// ---------------------------------------------------------------------------
// SA1 grouping: centered+normalized xyz only, zero-padded to CinPad (f16,
// pixel-major activation layout for WMMA B-fragment loads).
// ---------------------------------------------------------------------------
__global__ void group_xyz_kernel(const float* __restrict__ xyz, const float* __restrict__ cen,
                                 const int* __restrict__ idx, _Float16* __restrict__ act,
                                 int P, int N, int S, int K, int CinPad, float invR) {
  int pixel = blockIdx.x * blockDim.x + threadIdx.x;
  if (pixel >= P) return;
  int s = (pixel / K) % S;
  int b = pixel / (K * S);
  int n = idx[pixel];
  const float* pt = xyz + ((size_t)b * N + n) * 3;
  const float* c  = cen + ((size_t)b * S + s) * 3;
  _Float16* o = act + (size_t)pixel * CinPad;
  o[0] = (_Float16)((pt[0]-c[0]) * invR);
  o[1] = (_Float16)((pt[1]-c[1]) * invR);
  o[2] = (_Float16)((pt[2]-c[2]) * invR);
  for (int j = 3; j < CinPad; ++j) o[j] = (_Float16)0.f;
}

// SA2/SA4 grouping: normalized xyz + gathered features (B,C,N), zero-padded.
__global__ void group_feat_kernel(const float* __restrict__ xyz, const float* __restrict__ cen,
                                  const int* __restrict__ idx, const float* __restrict__ feats,
                                  _Float16* __restrict__ act,
                                  int P, int N, int S, int K, int C, int CinPad, float invR) {
  int pixel = blockIdx.x * blockDim.y + threadIdx.y;
  if (pixel >= P) return;
  int lane = threadIdx.x;
  int s = (pixel / K) % S;
  int b = pixel / (K * S);
  int n = idx[pixel];
  _Float16* o = act + (size_t)pixel * CinPad;
  if (lane < 3) {
    float v = (xyz[((size_t)b*N+n)*3 + lane] - cen[((size_t)b*S+s)*3 + lane]) * invR;
    o[lane] = (_Float16)v;
  }
  for (int cc = lane; cc < CinPad - 3; cc += 32) {
    float v = (cc < C) ? feats[((size_t)b*C + cc)*N + n] : 0.f;
    o[3 + cc] = (_Float16)v;
  }
}

// SA3 group-all: raw xyz + features, no centering/normalization.
__global__ void group_all_kernel(const float* __restrict__ xyz, const float* __restrict__ feats,
                                 _Float16* __restrict__ act, int P, int N, int C, int CinPad) {
  int pixel = blockIdx.x * blockDim.y + threadIdx.y;   // pixel = b*N + n
  if (pixel >= P) return;
  int lane = threadIdx.x;
  int n = pixel % N;
  int b = pixel / N;
  _Float16* o = act + (size_t)pixel * CinPad;
  if (lane < 3) o[lane] = (_Float16)xyz[(size_t)pixel*3 + lane];
  for (int cc = lane; cc < CinPad - 3; cc += 32) {
    float v = (cc < C) ? feats[((size_t)b*C + cc)*N + n] : 0.f;
    o[3 + cc] = (_Float16)v;
  }
}

// Pack f32 weights (Cout,Cin) -> f16 (Cout,CinPad), zero-padding K.
__global__ void pack_w_kernel(const float* __restrict__ W, _Float16* __restrict__ Wh,
                              int Cout, int Cin, int CinPad) {
  int i = blockIdx.x * blockDim.x + threadIdx.x;
  if (i >= Cout * CinPad) return;
  int o = i / CinPad, c = i % CinPad;
  Wh[i] = (_Float16)((c < Cin) ? W[(size_t)o*Cin + c] : 0.f);
}

// ---------------------------------------------------------------------------
// WMMA MLP layer: out(P,Cout) = relu(scale*(W@actT + bias) + beta), f16 I/O,
// f32 accumulate via v_wmma_f32_16x16x32_f16. One wave per 16x16 output tile.
// A-frag (ISA 16-bit A 16x32): lane row = lane&15, halves 0-7 <- K kbase..+7,
// halves 8-15 <- K kbase+16..+23, kbase = 8*(lane>=16). B-frag: lane col =
// lane&15, same K order as A. D: VGPR r -> row r+8*(lane>=16).
// ---------------------------------------------------------------------------
__global__ void wmma_layer_kernel(const _Float16* __restrict__ act, const _Float16* __restrict__ W,
                                  const float* __restrict__ bias, const float* __restrict__ gamma,
                                  const float* __restrict__ beta, _Float16* __restrict__ out,
                                  int P, int CinPad, int Cout) {
  int tileN = blockIdx.x * blockDim.y + threadIdx.y;
  int tileM = blockIdx.y;
  int lane  = threadIdx.x;
  int hi    = lane >> 4;        // 0 | 1
  int l     = lane & 15;
  const _Float16* wrow = W   + (size_t)(tileM * 16 + l) * CinPad;
  const _Float16* arow = act + (size_t)(tileN * 16 + l) * CinPad;
  int akOff = hi * 8;
  int bkOff = hi * 8;
  v8f acc = {};
  for (int k0 = 0; k0 < CinPad; k0 += 32) {
    v8h alo = *(const v8h*)(wrow + k0 + akOff);
    v8h ahi = *(const v8h*)(wrow + k0 + akOff + 16);
    v16h a = __builtin_shufflevector(alo, ahi, 0,1,2,3,4,5,6,7,8,9,10,11,12,13,14,15);
    v8h blo = *(const v8h*)(arow + k0 + bkOff);
    v8h bhi = *(const v8h*)(arow + k0 + bkOff + 16);
    v16h bm = __builtin_shufflevector(blo, bhi, 0,1,2,3,4,5,6,7,8,9,10,11,12,13,14,15);
    acc = __builtin_amdgcn_wmma_f32_16x16x32_f16(false, a, false, bm,
                                                 (short)0, acc, false, false);
  }
  const float sdenom = rsqrtf(1.f + BN_EPS);
  int ocBase = tileM * 16 + hi * 8;
  v8h ov;
#pragma unroll
  for (int r = 0; r < 8; ++r) {
    int oc = ocBase + r;
    float v = gamma[oc] * sdenom * (acc[r] + bias[oc]) + beta[oc];
    ov[r] = (_Float16)fmaxf(v, 0.f);
  }
  int pixel = tileN * 16 + l;
  *(v8h*)(out + (size_t)pixel * Cout + ocBase) = ov;   // 8 contiguous channels
}

// Max over nsample; act (B*S*K, C) pixel-major -> out (B,C,S) f32.
__global__ void maxpool_kernel(const _Float16* __restrict__ act, float* __restrict__ out,
                               int B, int S, int K, int C) {
  int i = blockIdx.x * blockDim.x + threadIdx.x;
  if (i >= B*S*C) return;
  int c = i % C; int s = (i / C) % S; int b = i / (C * S);
  const _Float16* base = act + ((size_t)(b*S + s) * K) * C + c;
  float m = -3.0e38f;
  for (int k = 0; k < K; ++k) m = fmaxf(m, (float)base[(size_t)k * C]);
  out[((size_t)b*C + c)*S + s] = m;
}

// ---------------------------------------------------------------------------
extern "C" void kernel_launch(void* const* d_in, const int* in_sizes, int n_in,
                              void* d_out, int out_size, void* d_ws, size_t ws_size,
                              hipStream_t stream) {
  (void)in_sizes; (void)n_in; (void)out_size; (void)ws_size;
  const int B = 16, N0 = 20000;
  const float* pclist = (const float*)d_in[0];
  auto pIn = [&](int stage, int layer, int which) -> const float* {
    return (const float*)d_in[1 + stage*12 + layer*4 + which];  // W,b,g,beta
  };

  size_t off = 0;
  auto alloc = [&](size_t bytes) -> char* {
    char* p = (char*)d_ws + off;
    off += (bytes + 255) & ~(size_t)255;
    return p;
  };
  float* dist  = (float*)alloc((size_t)B*N0*sizeof(float));
  int* fpsIdx1 = (int*)alloc((size_t)B*256*sizeof(int));
  int* fpsIdx2 = (int*)alloc((size_t)B*64*sizeof(int));
  int* fpsIdx4 = (int*)alloc((size_t)B*32*sizeof(int));
  float* xyz1  = (float*)alloc((size_t)B*256*3*sizeof(float));
  float* xyz2  = (float*)alloc((size_t)B*64*3*sizeof(float));
  float* xyz4  = (float*)alloc((size_t)B*32*3*sizeof(float));
  int* ball1   = (int*)alloc((size_t)B*256*32*sizeof(int));
  int* ball2   = (int*)alloc((size_t)B*64*64*sizeof(int));
  int* ball4   = (int*)alloc((size_t)B*32*64*sizeof(int));
  float* f1    = (float*)alloc((size_t)B*128*256*sizeof(float));
  float* f2    = (float*)alloc((size_t)B*256*64*sizeof(float));

  static const int cfg[4][3][3] = {   // {Cin, CinPad, Cout}
    {{  3, 32, 64},{ 64, 64, 64},{ 64, 64,128}},
    {{131,160,128},{128,128,128},{128,128,256}},
    {{259,288,128},{128,128,128},{128,128,256}},
    {{259,288,128},{128,128,128},{128,128,256}},
  };
  _Float16* wh[4][3];
  for (int s = 0; s < 4; ++s)
    for (int l = 0; l < 3; ++l)
      wh[s][l] = (_Float16*)alloc((size_t)cfg[s][l][1]*cfg[s][l][2]*sizeof(_Float16));

  const size_t ACT_ELEMS = (size_t)131072 * 128;   // largest activation
  _Float16* actA = (_Float16*)alloc(ACT_ELEMS * sizeof(_Float16));
  _Float16* actB = (_Float16*)alloc(ACT_ELEMS * sizeof(_Float16));

  for (int s = 0; s < 4; ++s)
    for (int l = 0; l < 3; ++l) {
      int tot = cfg[s][l][1]*cfg[s][l][2];
      pack_w_kernel<<<(tot+255)/256, 256, 0, stream>>>(pIn(s,l,0), wh[s][l],
                                                       cfg[s][l][2], cfg[s][l][0], cfg[s][l][1]);
    }

  auto layer = [&](int s, int l, const _Float16* in, _Float16* outp, int P) {
    int CinPad = cfg[s][l][1], Cout = cfg[s][l][2];
    dim3 grid(P/64, Cout/16), block(32, 4);   // 4 waves/block, 1 tile per wave
    wmma_layer_kernel<<<grid, block, 0, stream>>>(in, wh[s][l], pIn(s,l,1),
                                                  pIn(s,l,2), pIn(s,l,3), outp,
                                                  P, CinPad, Cout);
  };

  float* outF3 = (float*)d_out;               // (B,256,1)
  float* outF4 = (float*)d_out + B*256;       // (B,256,32)

  // ---- SA1: 20000 -> 256 centers, K=32, MLP 3->64->64->128 ----
  {
    int S = 256, K = 32, P = B*S*K;
    fps_kernel<<<B, 256, 0, stream>>>(pclist, fpsIdx1, dist, N0, S);
    gather_xyz_kernel<<<(B*S+255)/256, 256, 0, stream>>>(pclist, fpsIdx1, xyz1, B*S, N0, S);
    { dim3 g((B*S+7)/8), blk(32,8);
      ball_query_kernel<<<g, blk, 0, stream>>>(pclist, xyz1, ball1, B*S, N0, S, K, 0.2f*0.2f); }
    group_xyz_kernel<<<(P+255)/256, 256, 0, stream>>>(pclist, xyz1, ball1, actA,
                                                      P, N0, S, K, 32, 1.f/0.2f);
    layer(0,0, actA, actB, P);
    layer(0,1, actB, actA, P);
    layer(0,2, actA, actB, P);
    maxpool_kernel<<<(B*S*128+255)/256, 256, 0, stream>>>(actB, f1, B, S, K, 128);
  }
  // ---- SA2: 256 -> 64 centers, K=64, MLP 131->128->128->256 ----
  {
    int N = 256, S = 64, K = 64, P = B*S*K;
    fps_kernel<<<B, 256, 0, stream>>>(xyz1, fpsIdx2, dist, N, S);
    gather_xyz_kernel<<<(B*S+255)/256, 256, 0, stream>>>(xyz1, fpsIdx2, xyz2, B*S, N, S);
    { dim3 g((B*S+7)/8), blk(32,8);
      ball_query_kernel<<<g, blk, 0, stream>>>(xyz1, xyz2, ball2, B*S, N, S, K, 0.4f*0.4f); }
    { dim3 g((P+7)/8), blk(32,8);
      group_feat_kernel<<<g, blk, 0, stream>>>(xyz1, xyz2, ball2, f1, actA,
                                               P, N, S, K, 128, 160, 1.f/0.4f); }
    layer(1,0, actA, actB, P);
    layer(1,1, actB, actA, P);
    layer(1,2, actA, actB, P);
    maxpool_kernel<<<(B*S*256+255)/256, 256, 0, stream>>>(actB, f2, B, S, K, 256);
  }
  // ---- SA3: group-all over 64 pts, MLP 259->128->128->256, out (B,256,1) ----
  {
    int N = 64, P = B*N;   // 1024 pixels
    { dim3 g((P+7)/8), blk(32,8);
      group_all_kernel<<<g, blk, 0, stream>>>(xyz2, f2, actA, P, N, 256, 288); }
    layer(2,0, actA, actB, P);
    layer(2,1, actB, actA, P);
    layer(2,2, actA, actB, P);
    maxpool_kernel<<<(B*256+255)/256, 256, 0, stream>>>(actB, outF3, B, 1, N, 256);
  }
  // ---- SA4: 64 -> 32 centers, K=64, MLP 259->128->128->256, out (B,256,32) ----
  {
    int N = 64, S = 32, K = 64, P = B*S*K;
    fps_kernel<<<B, 256, 0, stream>>>(xyz2, fpsIdx4, dist, N, S);
    gather_xyz_kernel<<<(B*S+255)/256, 256, 0, stream>>>(xyz2, fpsIdx4, xyz4, B*S, N, S);
    { dim3 g((B*S+7)/8), blk(32,8);
      ball_query_kernel<<<g, blk, 0, stream>>>(xyz2, xyz4, ball4, B*S, N, S, K, 0.6f*0.6f); }
    { dim3 g((P+7)/8), blk(32,8);
      group_feat_kernel<<<g, blk, 0, stream>>>(xyz2, xyz4, ball4, f2, actA,
                                               P, N, S, K, 256, 288, 1.f/0.6f); }
    layer(3,0, actA, actB, P);
    layer(3,1, actB, actA, P);
    layer(3,2, actA, actB, P);
    maxpool_kernel<<<(B*S*256+255)/256, 256, 0, stream>>>(actB, outF4, B, S, K, 256);
  }
}
